// HeteroMPNNLayer_45896020525884
// MI455X (gfx1250) — matrix operand order repaired and checked by reading, not compiled, not run on hardware
//
#include <hip/hip_runtime.h>

#define H      128
#define NNODES 100000
#define ECNT   250000
#define KT     4

typedef __attribute__((ext_vector_type(16))) __bf16 v16bf;
typedef __attribute__((ext_vector_type(8)))  __bf16 v8bf;
typedef __attribute__((ext_vector_type(8)))  float  v8f;

#define EWAVES 8          // waves (16-row tiles) per workgroup
#define TPB    (EWAVES * 32)
#define EDGE_WGS 128      // workgroups per edge type
#define NODE_WGS 320

// ---------------------------------------------------------------------------
// Convert a row-major f32 weight matrix [Kdim][Ncols] to bf16 and store it in
// LDS pre-swizzled into WMMA B-fragment order:
//   fragment(kc,nt) occupies 32 lanes * 16 bf16 (32B per lane, contiguous),
//   lane = n%16 + (((k%32)>>3)&1)*16, elem = (k%8) + 8*((k%32)>>4)   (B 32x16 layout).
// Each B operand load is then a single contiguous 32B read per lane.
// ---------------------------------------------------------------------------
__device__ inline void fill_swizzled(const float* __restrict__ Wg, __bf16* Ws,
                                     int Kdim, int Ncols, int tid, int nthreads) {
  const int NT = Ncols >> 4;
  const int n4cnt = Ncols >> 2;
  const int total4 = (Kdim * Ncols) >> 2;
  for (int idx = tid; idx < total4; idx += nthreads) {
    int K  = idx / n4cnt;
    int n4 = (idx - K * n4cnt) << 2;
    float4 w = *(const float4*)(Wg + (size_t)K * Ncols + n4);
    int kc = K >> 5;
    int kin = K & 31;
    int elem = (kin & 7) | ((kin >> 4) << 3);
    int laneHalf = ((kin >> 3) & 1) << 4;
    float wv[4] = {w.x, w.y, w.z, w.w};
#pragma unroll
    for (int q = 0; q < 4; ++q) {
      int n = n4 + q;
      int nt = n >> 4, nin = n & 15;
      Ws[(size_t)((kc * NT + nt) * 32 + (nin + laneHalf)) * 16 + elem] = (__bf16)wv[q];
    }
  }
}

__device__ inline v16bf load_bfrag(const __bf16* Ws, int NT, int kc, int nt, int lane) {
  return *(const v16bf*)(Ws + (size_t)((kc * NT + nt) * 32 + lane) * 16);
}

// A fragment (16x32 bf16) assembled from two aligned 8-f32 runs (float4 pairs).
// lane<16: elems 0-7 = K 0..7,  8-15 = K 16..23 (of this k-chunk)
// lane>=16: elems 0-7 = K 8..15, 8-15 = K 24..31
__device__ inline v16bf loadA_f32(const float* p1, const float* p2) {
  float4 x0 = ((const float4*)p1)[0];
  float4 x1 = ((const float4*)p1)[1];
  float4 y0 = ((const float4*)p2)[0];
  float4 y1 = ((const float4*)p2)[1];
  v16bf a;
  a[0]=(__bf16)x0.x;  a[1]=(__bf16)x0.y;  a[2]=(__bf16)x0.z;  a[3]=(__bf16)x0.w;
  a[4]=(__bf16)x1.x;  a[5]=(__bf16)x1.y;  a[6]=(__bf16)x1.z;  a[7]=(__bf16)x1.w;
  a[8]=(__bf16)y0.x;  a[9]=(__bf16)y0.y;  a[10]=(__bf16)y0.z; a[11]=(__bf16)y0.w;
  a[12]=(__bf16)y1.x; a[13]=(__bf16)y1.y; a[14]=(__bf16)y1.z; a[15]=(__bf16)y1.w;
  return a;
}

// Same fragment pattern but from a row-major bf16 LDS buffer.
__device__ inline v16bf loadA_lds(const __bf16* base) {
  v8bf lo = *(const v8bf*)base;
  v8bf hi = *(const v8bf*)(base + 16);
  v16bf a;
#pragma unroll
  for (int i = 0; i < 8; ++i) { a[i] = lo[i]; a[8 + i] = hi[i]; }
  return a;
}

__device__ inline v8f wmma_bf16(v16bf a, v16bf b, v8f c) {
  return __builtin_amdgcn_wmma_f32_16x16x32_bf16(false, a, false, b, (short)0, c, false, false);
}

// ---------------------------------------------------------------------------
// Zero the aggregation buffer (d_out is reused as the segment-sum accumulator).
// ---------------------------------------------------------------------------
__global__ void zero_kernel(float* __restrict__ p, int n4) {
  float4 z = {0.f, 0.f, 0.f, 0.f};
  for (int i = blockIdx.x * blockDim.x + threadIdx.x; i < n4; i += gridDim.x * blockDim.x)
    ((float4*)p)[i] = z;
}

// ---------------------------------------------------------------------------
// Edge MLP + scatter-add.  blockIdx.y = edge type.  Each wave owns a 16-edge
// tile: m = ReLU([h_src|h_dst|ee] @ W1 + b1) @ W2 + b2, atomically added to agg.
// ---------------------------------------------------------------------------
__global__ void __launch_bounds__(TPB) edge_mlp_kernel(
    const float* __restrict__ h, const long long* __restrict__ ei,
    const float* __restrict__ ee, const float* __restrict__ eW1,
    const float* __restrict__ eb1, const float* __restrict__ eW2,
    const float* __restrict__ eb2, float* __restrict__ agg) {
  __shared__ __align__(32) __bf16 W1s[384 * 256];          // 192 KB
  __shared__ __align__(32) __bf16 W2s[256 * 128];          //  64 KB
  __shared__ __align__(32) __bf16 zbuf[EWAVES][16 * 256];  //  64 KB  (320 KB total)

  const int ktype = blockIdx.y;
  const int tid = threadIdx.x;
  fill_swizzled(eW1 + (size_t)ktype * 384 * 256, W1s, 384, 256, tid, TPB);
  fill_swizzled(eW2 + (size_t)ktype * 256 * 128, W2s, 256, 128, tid, TPB);
  __syncthreads();

  const int wid = tid >> 5;
  const int lane = tid & 31;
  const int r = lane & 15;        // tile row handled by this lane for A frags
  const int halfhi = lane >> 4;   // 0: low K half, 1: high K half
  const int s8 = halfhi << 3;
  const int col0 = lane & 15;     // output column within 16-wide N tile

  const long long* src_g = ei + (size_t)ktype * 2 * ECNT;
  const long long* dst_g = src_g + ECNT;
  const float* ee_k = ee + (size_t)ktype * ECNT * H;
  const int ntiles = ECNT / 16;   // 15625, exact
  __bf16* zw = &zbuf[wid][0];

  // Tile-invariant per-lane bias values.
  float bias1v[16], bias2v[8];
#pragma unroll
  for (int nt = 0; nt < 16; ++nt) bias1v[nt] = eb1[(size_t)ktype * 256 + nt * 16 + col0];
#pragma unroll
  for (int nt = 0; nt < 8; ++nt)  bias2v[nt] = eb2[(size_t)ktype * 128 + nt * 16 + col0];

  for (int t0 = blockIdx.x * EWAVES; t0 < ntiles; t0 += gridDim.x * EWAVES) {
    int tile = t0 + wid;
    bool valid = tile < ntiles;
    int tt = valid ? tile : ntiles - 1;   // clamp; stores are predicated below
    int e0 = tt << 4;

    const float* psrc = h + (size_t)src_g[e0 + r] * H;
    const float* pdst = h + (size_t)dst_g[e0 + r] * H;
    const float* pee  = ee_k + (size_t)(e0 + r) * H;

    // Layer-1 A fragments (16 x 384) straight from global gathers, kept in regs.
    v16bf A1[12];
#pragma unroll
    for (int kc = 0; kc < 12; ++kc) {
      int c1 = kc * 32 + s8;
      int c2 = c1 + 16;
      const float* p1 = (c1 < 128) ? (psrc + c1) : (c1 < 256) ? (pdst + c1 - 128) : (pee + c1 - 256);
      const float* p2 = (c2 < 128) ? (psrc + c2) : (c2 < 256) ? (pdst + c2 - 128) : (pee + c2 - 256);
      A1[kc] = loadA_f32(p1, p2);
    }
    __builtin_amdgcn_sched_barrier(0);   // retire f32 staging before the MMA loop

    // z = ReLU(A @ W1 + b1); two independent accumulator chains per pass.
#pragma unroll 1
    for (int nt = 0; nt < 16; nt += 2) {
      v8f c0 = {}, c1a = {};
#pragma unroll
      for (int kc = 0; kc < 12; ++kc) {
        v16bf b0 = load_bfrag(W1s, 16, kc, nt, lane);
        v16bf b1 = load_bfrag(W1s, 16, kc, nt + 1, lane);
        c0  = wmma_bf16(A1[kc], b0, c0);
        c1a = wmma_bf16(A1[kc], b1, c1a);
      }
#pragma unroll
      for (int j = 0; j < 8; ++j) {
        float v0 = c0[j]  + bias1v[nt];
        float v1 = c1a[j] + bias1v[nt + 1];
        v0 = v0 > 0.f ? v0 : 0.f;
        v1 = v1 > 0.f ? v1 : 0.f;
        zw[(j + halfhi * 8) * 256 + nt * 16 + col0]       = (__bf16)v0;
        zw[(j + halfhi * 8) * 256 + (nt + 1) * 16 + col0] = (__bf16)v1;
      }
    }
    __syncthreads();

    v16bf A2[8];
#pragma unroll
    for (int kc = 0; kc < 8; ++kc)
      A2[kc] = loadA_lds(zw + r * 256 + kc * 32 + s8);
    __builtin_amdgcn_sched_barrier(0);

    int dj[8];
#pragma unroll
    for (int j = 0; j < 8; ++j)
      dj[j] = (int)dst_g[e0 + j + halfhi * 8];

    // m = z @ W2 + b2 ; scatter-add into agg (L2-resident atomics).
#pragma unroll 1
    for (int nt = 0; nt < 8; nt += 2) {
      v8f c0 = {}, c1a = {};
#pragma unroll
      for (int kc = 0; kc < 8; ++kc) {
        v16bf b0 = load_bfrag(W2s, 8, kc, nt, lane);
        v16bf b1 = load_bfrag(W2s, 8, kc, nt + 1, lane);
        c0  = wmma_bf16(A2[kc], b0, c0);
        c1a = wmma_bf16(A2[kc], b1, c1a);
      }
      if (valid) {
#pragma unroll
        for (int j = 0; j < 8; ++j) {
          __hip_atomic_fetch_add(&agg[(size_t)dj[j] * H + nt * 16 + col0],
                                 c0[j] + bias2v[nt],
                                 __ATOMIC_RELAXED, __HIP_MEMORY_SCOPE_AGENT);
          __hip_atomic_fetch_add(&agg[(size_t)dj[j] * H + (nt + 1) * 16 + col0],
                                 c1a[j] + bias2v[nt + 1],
                                 __ATOMIC_RELAXED, __HIP_MEMORY_SCOPE_AGENT);
        }
      }
    }
    __syncthreads();
  }
}

// ---------------------------------------------------------------------------
// Node MLP + residual + LayerNorm.  agg aliases out (read-before-write per tile).
// ---------------------------------------------------------------------------
__global__ void __launch_bounds__(TPB) node_kernel(
    const float* __restrict__ h, const float* agg,
    const float* __restrict__ nW1, const float* __restrict__ nb1,
    const float* __restrict__ nW2, const float* __restrict__ nb2,
    const float* __restrict__ gamma, const float* __restrict__ beta,
    float* out) {
  __shared__ __align__(32) __bf16 W1s[256 * 256];          // 128 KB
  __shared__ __align__(32) __bf16 W2s[256 * 128];          //  64 KB
  __shared__ __align__(32) __bf16 zbuf[EWAVES][16 * 256];  //  64 KB
  __shared__ __align__(16) float  ubuf[EWAVES][16 * 128];  //  64 KB  (320 KB total)

  const int tid = threadIdx.x;
  fill_swizzled(nW1, W1s, 256, 256, tid, TPB);
  fill_swizzled(nW2, W2s, 256, 128, tid, TPB);
  __syncthreads();

  const int wid = tid >> 5;
  const int lane = tid & 31;
  const int r = lane & 15;
  const int halfhi = lane >> 4;
  const int s8 = halfhi << 3;
  const int col0 = lane & 15;
  const int ntiles = NNODES / 16;   // 6250, exact
  __bf16* zw = &zbuf[wid][0];
  float*  uw = &ubuf[wid][0];

  float bias1v[16], bias2v[8];
#pragma unroll
  for (int nt = 0; nt < 16; ++nt) bias1v[nt] = nb1[nt * 16 + col0];
#pragma unroll
  for (int nt = 0; nt < 8; ++nt)  bias2v[nt] = nb2[nt * 16 + col0];

  for (int t0 = blockIdx.x * EWAVES; t0 < ntiles; t0 += gridDim.x * EWAVES) {
    int tile = t0 + wid;
    bool valid = tile < ntiles;
    int tt = valid ? tile : ntiles - 1;
    int row0 = tt << 4;

    const float* ph = h   + (size_t)(row0 + r) * H;
    const float* pa = agg + (size_t)(row0 + r) * H;

    v16bf A1[8];
#pragma unroll
    for (int kc = 0; kc < 8; ++kc) {
      int c1 = kc * 32 + s8;
      int c2 = c1 + 16;
      const float* p1 = (c1 < 128) ? (ph + c1) : (pa + c1 - 128);
      const float* p2 = (c2 < 128) ? (ph + c2) : (pa + c2 - 128);
      A1[kc] = loadA_f32(p1, p2);
    }
    __builtin_amdgcn_sched_barrier(0);

#pragma unroll 1
    for (int nt = 0; nt < 16; nt += 2) {
      v8f c0 = {}, c1a = {};
#pragma unroll
      for (int kc = 0; kc < 8; ++kc) {
        v16bf b0 = load_bfrag(W1s, 16, kc, nt, lane);
        v16bf b1 = load_bfrag(W1s, 16, kc, nt + 1, lane);
        c0  = wmma_bf16(A1[kc], b0, c0);
        c1a = wmma_bf16(A1[kc], b1, c1a);
      }
#pragma unroll
      for (int j = 0; j < 8; ++j) {
        float v0 = c0[j]  + bias1v[nt];
        float v1 = c1a[j] + bias1v[nt + 1];
        v0 = v0 > 0.f ? v0 : 0.f;
        v1 = v1 > 0.f ? v1 : 0.f;
        zw[(j + halfhi * 8) * 256 + nt * 16 + col0]       = (__bf16)v0;
        zw[(j + halfhi * 8) * 256 + (nt + 1) * 16 + col0] = (__bf16)v1;
      }
    }
    __syncthreads();

    v16bf A2[8];
#pragma unroll
    for (int kc = 0; kc < 8; ++kc)
      A2[kc] = loadA_lds(zw + r * 256 + kc * 32 + s8);
    __builtin_amdgcn_sched_barrier(0);

#pragma unroll 1
    for (int nt = 0; nt < 8; nt += 2) {
      v8f c0 = {}, c1a = {};
#pragma unroll
      for (int kc = 0; kc < 8; ++kc) {
        v16bf b0 = load_bfrag(W2s, 8, kc, nt, lane);
        v16bf b1 = load_bfrag(W2s, 8, kc, nt + 1, lane);
        c0  = wmma_bf16(A2[kc], b0, c0);
        c1a = wmma_bf16(A2[kc], b1, c1a);
      }
#pragma unroll
      for (int j = 0; j < 8; ++j) {
        uw[(j + halfhi * 8) * 128 + nt * 16 + col0]       = c0[j]  + bias2v[nt];
        uw[(j + halfhi * 8) * 128 + (nt + 1) * 16 + col0] = c1a[j] + bias2v[nt + 1];
      }
    }
    __syncthreads();

    // Residual + LayerNorm: lanes 0..15 each own one row.
    if (valid && lane < 16) {
      const float* hrow = h + (size_t)(row0 + lane) * H;
      const float* urow = uw + lane * 128;
      float s = 0.f, ss = 0.f;
      for (int c4 = 0; c4 < 128; c4 += 4) {
        float4 hv = *(const float4*)(hrow + c4);
        float4 uv = *(const float4*)(urow + c4);
        float x0 = hv.x + uv.x, x1 = hv.y + uv.y, x2 = hv.z + uv.z, x3 = hv.w + uv.w;
        s  += x0 + x1 + x2 + x3;
        ss += x0 * x0 + x1 * x1 + x2 * x2 + x3 * x3;
      }
      float mu  = s * (1.f / 128.f);
      float var = ss * (1.f / 128.f) - mu * mu;
      float rs  = rsqrtf(var + 1e-5f);
      float* orow = out + (size_t)(row0 + lane) * H;
      for (int c4 = 0; c4 < 128; c4 += 4) {
        float4 hv = *(const float4*)(hrow + c4);
        float4 uv = *(const float4*)(urow + c4);
        float4 gv = *(const float4*)(gamma + c4);
        float4 bv = *(const float4*)(beta + c4);
        float4 o;
        o.x = (hv.x + uv.x - mu) * rs * gv.x + bv.x;
        o.y = (hv.y + uv.y - mu) * rs * gv.y + bv.y;
        o.z = (hv.z + uv.z - mu) * rs * gv.z + bv.z;
        o.w = (hv.w + uv.w - mu) * rs * gv.w + bv.w;
        *(float4*)(orow + c4) = o;
      }
    }
    __syncthreads();
  }
}

extern "C" void kernel_launch(void* const* d_in, const int* in_sizes, int n_in,
                              void* d_out, int out_size, void* d_ws, size_t ws_size,
                              hipStream_t stream) {
  (void)in_sizes; (void)n_in; (void)out_size; (void)d_ws; (void)ws_size;
  const float*     h     = (const float*)d_in[0];
  const long long* ei    = (const long long*)d_in[1];
  const float*     ee    = (const float*)d_in[2];
  const float*     eW1   = (const float*)d_in[3];
  const float*     eb1   = (const float*)d_in[4];
  const float*     eW2   = (const float*)d_in[5];
  const float*     eb2   = (const float*)d_in[6];
  const float*     nW1   = (const float*)d_in[7];
  const float*     nb1   = (const float*)d_in[8];
  const float*     nW2   = (const float*)d_in[9];
  const float*     nb2   = (const float*)d_in[10];
  const float*     gamma = (const float*)d_in[11];
  const float*     beta  = (const float*)d_in[12];
  float* out = (float*)d_out;
  float* agg = out;   // reuse output buffer as the segment-sum accumulator

  zero_kernel<<<256, 256, 0, stream>>>(agg, NNODES * H / 4);
  dim3 egrid(EDGE_WGS, KT);
  edge_mlp_kernel<<<egrid, TPB, 0, stream>>>(h, ei, ee, eW1, eb1, eW2, eb2, agg);
  node_kernel<<<NODE_WGS, TPB, 0, stream>>>(h, agg, nW1, nb1, nW2, nb2, gamma, beta, out);
}
